// DeMIE_GATv2_6116033429989
// MI455X (gfx1250) — matrix operand order repaired and checked by reading, not compiled, not run on hardware
//
#include <hip/hip_runtime.h>
#include <stdint.h>
#include <math.h>

#define HEADS 4
#define HID   32
#define FIN   128  // IN_C == HEADS*HID

typedef __attribute__((ext_vector_type(16))) __bf16 v16bf;
typedef __attribute__((ext_vector_type(8)))  float  v8f;

__device__ __forceinline__ float lrelu(float x, float s){ return fmaxf(x, s*x); }

// float atomic max via int/uint punning (deterministic, native atomics)
__device__ __forceinline__ void atomicMaxF(float* addr, float v){
  if (v >= 0.f) atomicMax((int*)addr, __float_as_int(v));
  else          atomicMin((unsigned int*)addr, (unsigned int)__float_as_int(v));
}

__device__ __forceinline__ unsigned f2bf(float f){
  __bf16 h = (__bf16)f;
  unsigned short u; __builtin_memcpy(&u, &h, 2);
  return (unsigned)u;
}

// ---------------- fill ----------------
__global__ void k_fill(float* p, float v, long long n){
  long long i = (long long)blockIdx.x*blockDim.x + threadIdx.x;
  if (i < n) p[i] = v;
}

// ------- pack W [K x NC] f32 row-major -> WMMA-B bf16 layout -------
// Loader (lane group g=lane>>4, column n): 8 u32 at ((kblk*NC+n)*2+g)*8,
// u32 i = {W[k+1][n],W[k][n]}, k = kblk*32 + 8*g + 2*(i&3) + 16*(i>>2).
__global__ void k_pack_b(const float* __restrict__ W, uint32_t* __restrict__ pk,
                         int K, int NC){
  int t = blockIdx.x*blockDim.x + threadIdx.x;
  int total = (K/32)*NC*16;
  if (t >= total) return;
  int i = t & 7, g = (t>>3)&1;
  int n = (t>>4) % NC;
  int kblk = t / (16*NC);
  int k = kblk*32 + 8*g + 2*(i&3) + 16*(i>>2);
  unsigned lo = f2bf(W[(size_t)k*NC + n]);
  unsigned hi = f2bf(W[(size_t)(k+1)*NC + n]);
  pk[t] = (hi<<16) | lo;
}

// ------- dual WMMA GEMM: O1 = X@W1, O2 = X@W2; X [nrows x 128] f32 -------
// block = 2*(NC/16) waves; wave -> (matrix, col-tile); 16-row strip per block.
__global__ void k_gemm_dual(const float* __restrict__ X,
                            const uint32_t* __restrict__ p1,
                            const uint32_t* __restrict__ p2,
                            float* __restrict__ O1, float* __restrict__ O2,
                            int nrows, int NC){
  const int lane  = threadIdx.x & 31;
  const int wave  = threadIdx.x >> 5;
  const int ntile = NC >> 4;
  const int mat   = (wave >= ntile);
  const int ct    = mat ? wave - ntile : wave;
  const uint32_t* __restrict__ pk = mat ? p2 : p1;
  float* __restrict__ O = mat ? O2 : O1;
  const int m0 = blockIdx.x*16;
  int mrow = m0 + (lane & 15);
  if (mrow >= nrows) mrow = nrows - 1;        // clamp loads; stores guarded
  const int g    = lane >> 4;
  const int ncol = ct*16 + (lane & 15);
  const float* __restrict__ xrow = X + (size_t)mrow*128;
  v8f acc = {0.f,0.f,0.f,0.f,0.f,0.f,0.f,0.f};
#pragma unroll
  for (int kblk = 0; kblk < 4; ++kblk){
    // A 16x32 bf16 fragment: ks [kb0+8g .. +7] and [kb0+16+8g .. +7]
    const float* xp = xrow + kblk*32 + g*8;
    float4 a0 = *(const float4*)(xp);
    float4 a1 = *(const float4*)(xp+4);
    float4 a2 = *(const float4*)(xp+16);
    float4 a3 = *(const float4*)(xp+20);
    v16bf a;
    a[0]=(__bf16)a0.x;  a[1]=(__bf16)a0.y;  a[2]=(__bf16)a0.z;  a[3]=(__bf16)a0.w;
    a[4]=(__bf16)a1.x;  a[5]=(__bf16)a1.y;  a[6]=(__bf16)a1.z;  a[7]=(__bf16)a1.w;
    a[8]=(__bf16)a2.x;  a[9]=(__bf16)a2.y;  a[10]=(__bf16)a2.z; a[11]=(__bf16)a2.w;
    a[12]=(__bf16)a3.x; a[13]=(__bf16)a3.y; a[14]=(__bf16)a3.z; a[15]=(__bf16)a3.w;
    const uint32_t* bp = pk + ((size_t)(kblk*NC + ncol)*2 + g)*8;
    union { uint4 q[2]; v16bf v; } B;
    B.q[0] = *(const uint4*)(bp);
    B.q[1] = *(const uint4*)(bp+4);
    acc = __builtin_amdgcn_wmma_f32_16x16x32_bf16(false, a, false, B.v,
                                                  (short)0, acc, false, false);
  }
#pragma unroll
  for (int r = 0; r < 8; ++r){
    int row = m0 + g*8 + r;
    if (row < nrows) O[(size_t)row*NC + ncol] = acc[r];
  }
}

// ------- conv1 edge scores: e = sum_j att1[h][j]*lrelu(xl[s]+xr[d]) -------
__global__ void k_scores1(const float* __restrict__ xl, const float* __restrict__ xr,
                          const float* __restrict__ att1,
                          float* __restrict__ e1, float* __restrict__ m1,
                          const int* __restrict__ ei, int E, int Et){
  const int lane = threadIdx.x & 31;
  int ee = blockIdx.x*(blockDim.x>>5) + (threadIdx.x>>5);
  if (ee >= Et) return;
  int s, d;
  if (ee < E){ s = ei[ee]; d = ei[E + ee]; } else { s = d = ee - E; }
  const float* xs = xl + (size_t)s*FIN;
  const float* xd = xr + (size_t)d*FIN;
  float v[HEADS];
#pragma unroll
  for (int h = 0; h < HEADS; ++h){
    float t = xs[h*HID + lane] + xd[h*HID + lane];
    v[h] = att1[h*HID + lane] * lrelu(t, 0.2f);
  }
#pragma unroll
  for (int off = 16; off; off >>= 1)
#pragma unroll
    for (int h = 0; h < HEADS; ++h) v[h] += __shfl_xor(v[h], off, 32);
  if (lane == 0){
#pragma unroll
    for (int h = 0; h < HEADS; ++h){
      e1[(size_t)ee*HEADS + h] = v[h];
      atomicMaxF(&m1[d*HEADS + h], v[h]);
    }
  }
}

// ------- conv1 exp + segment sum; ex stored into alpha output region -------
__global__ void k_exp1(const float* __restrict__ e1, const float* __restrict__ m1,
                       float* __restrict__ s1, float* __restrict__ exo,
                       const int* __restrict__ ei, int E, int Et){
  int idx = blockIdx.x*blockDim.x + threadIdx.x;
  if (idx >= Et*HEADS) return;
  int ee = idx >> 2, h = idx & 3;
  int d = (ee < E) ? ei[E + ee] : ee - E;
  float ex = __expf(e1[idx] - m1[d*HEADS + h]);
  atomicAdd(&s1[d*HEADS + h], ex);
  exo[idx] = ex;
}

// ------- conv1 normalize (in place in d_out) + scatter aggregation -------
__global__ void k_agg1(float* __restrict__ alpha, const float* __restrict__ s1,
                       const float* __restrict__ xl, float* __restrict__ hacc,
                       const int* __restrict__ ei, int E, int Et){
  const int lane = threadIdx.x & 31;
  int ee = blockIdx.x*(blockDim.x>>5) + (threadIdx.x>>5);
  if (ee >= Et) return;
  int s, d;
  if (ee < E){ s = ei[ee]; d = ei[E + ee]; } else { s = d = ee - E; }
  float a = 0.f;
  if (lane < HEADS){
    float ex = alpha[(size_t)ee*HEADS + lane];
    a = ex / (s1[d*HEADS + lane] + 1e-16f);
    alpha[(size_t)ee*HEADS + lane] = a;     // final alpha output
  }
  const float* xs = xl + (size_t)s*FIN;
  float* hd = hacc + (size_t)d*FIN;
#pragma unroll
  for (int h = 0; h < HEADS; ++h){
    float ah = __shfl(a, h, 32);
    atomicAdd(&hd[h*HID + lane], ah * xs[h*HID + lane]);
  }
}

// ------- batchnorm stats per channel (incl. bias b1) -------
__global__ void k_bn_stats(const float* __restrict__ h1, const float* __restrict__ b1,
                           float* __restrict__ mu, float* __restrict__ rv, int n){
  const int c = blockIdx.x;
  const float bias = b1[c];
  float s = 0.f, s2 = 0.f;
  for (int i = threadIdx.x; i < n; i += blockDim.x){
    float v = h1[(size_t)i*FIN + c] + bias;
    s += v; s2 += v*v;
  }
  __shared__ float sh0[256], sh1[256];
  sh0[threadIdx.x] = s; sh1[threadIdx.x] = s2;
  __syncthreads();
  for (int o = 128; o > 0; o >>= 1){
    if ((int)threadIdx.x < o){ sh0[threadIdx.x] += sh0[threadIdx.x+o];
                               sh1[threadIdx.x] += sh1[threadIdx.x+o]; }
    __syncthreads();
  }
  if (threadIdx.x == 0){
    float m = sh0[0] / n;
    float var = sh1[0] / n - m*m;
    mu[c] = m;
    rv[c] = rsqrtf(var + 1e-5f);
  }
}

// ------- batchnorm apply + leaky_relu (in place on h1) -------
__global__ void k_bn_apply(float* __restrict__ h1, const float* __restrict__ b1,
                           const float* __restrict__ mu, const float* __restrict__ rv,
                           const float* __restrict__ gamma, const float* __restrict__ beta,
                           long long total){
  long long i = (long long)blockIdx.x*blockDim.x + threadIdx.x;
  if (i >= total) return;
  int c = (int)(i & (FIN-1));
  float v = h1[i] + b1[c];
  v = (v - mu[c]) * rv[c] * gamma[c] + beta[c];
  h1[i] = lrelu(v, 0.01f);
}

// ------- conv2 edge scores (1 head) -------
__global__ void k_scores2(const float* __restrict__ xl, const float* __restrict__ xr,
                          const float* __restrict__ att2,
                          float* __restrict__ e2, float* __restrict__ m2,
                          const int* __restrict__ ei, int E, int Et){
  const int lane = threadIdx.x & 31;
  int ee = blockIdx.x*(blockDim.x>>5) + (threadIdx.x>>5);
  if (ee >= Et) return;
  int s, d;
  if (ee < E){ s = ei[ee]; d = ei[E + ee]; } else { s = d = ee - E; }
  float t = xl[(size_t)s*HID + lane] + xr[(size_t)d*HID + lane];
  float v = att2[lane] * lrelu(t, 0.2f);
#pragma unroll
  for (int off = 16; off; off >>= 1) v += __shfl_xor(v, off, 32);
  if (lane == 0){
    e2[ee] = v;
    atomicMaxF(&m2[d], v);
  }
}

__global__ void k_exp2(float* __restrict__ e2, const float* __restrict__ m2,
                       float* __restrict__ s2, const int* __restrict__ ei,
                       int E, int Et){
  int ee = blockIdx.x*blockDim.x + threadIdx.x;
  if (ee >= Et) return;
  int d = (ee < E) ? ei[E + ee] : ee - E;
  float ex = __expf(e2[ee] - m2[d]);
  atomicAdd(&s2[d], ex);
  e2[ee] = ex;   // in place
}

__global__ void k_agg2(const float* __restrict__ e2, const float* __restrict__ s2,
                       const float* __restrict__ xl, float* __restrict__ h2,
                       const int* __restrict__ ei, int E, int Et){
  const int lane = threadIdx.x & 31;
  int ee = blockIdx.x*(blockDim.x>>5) + (threadIdx.x>>5);
  if (ee >= Et) return;
  int s, d;
  if (ee < E){ s = ei[ee]; d = ei[E + ee]; } else { s = d = ee - E; }
  float a = e2[ee] / (s2[d] + 1e-16f);
  atomicAdd(&h2[(size_t)d*HID + lane], a * xl[(size_t)s*HID + lane]);
}

// ------- final: bias + leaky, tiny GEMV to 3 logits, log_softmax -------
__global__ void k_out(const float* __restrict__ h2, const float* __restrict__ b2,
                      const float* __restrict__ Wout, const float* __restrict__ bout,
                      float* __restrict__ logp, int n){
  int i = blockIdx.x*blockDim.x + threadIdx.x;
  if (i >= n) return;
  float l0 = bout[0], l1 = bout[1], l2 = bout[2];
  const float* hv = h2 + (size_t)i*HID;
#pragma unroll
  for (int c = 0; c < HID; ++c){
    float v = lrelu(hv[c] + b2[c], 0.01f);
    l0 += v * Wout[c*3+0];
    l1 += v * Wout[c*3+1];
    l2 += v * Wout[c*3+2];
  }
  float m = fmaxf(l0, fmaxf(l1, l2));
  float ls = m + __logf(__expf(l0-m) + __expf(l1-m) + __expf(l2-m));
  logp[(size_t)i*3+0] = l0 - ls;
  logp[(size_t)i*3+1] = l1 - ls;
  logp[(size_t)i*3+2] = l2 - ls;
}

extern "C" void kernel_launch(void* const* d_in, const int* in_sizes, int n_in,
                              void* d_out, int out_size, void* d_ws, size_t ws_size,
                              hipStream_t stream){
  (void)n_in; (void)out_size; (void)ws_size;
  const float* x     = (const float*)d_in[0];
  const int*   ei    = (const int*)d_in[1];   // edge_index [2,E]
  const float* Wl1   = (const float*)d_in[2];
  const float* Wr1   = (const float*)d_in[3];
  const float* att1  = (const float*)d_in[4];
  const float* b1    = (const float*)d_in[5];
  const float* gamma = (const float*)d_in[6];
  const float* beta  = (const float*)d_in[7];
  const float* Wl2   = (const float*)d_in[8];
  const float* Wr2   = (const float*)d_in[9];
  const float* att2  = (const float*)d_in[10];
  const float* b2    = (const float*)d_in[11];
  const float* Wout  = (const float*)d_in[12];
  const float* bout  = (const float*)d_in[13];

  const int N  = in_sizes[0] / FIN;
  const int E  = in_sizes[1] / 2;
  const int Et = E + N;

  float* logp  = (float*)d_out;
  float* alpha = logp + (size_t)N*3;   // also used as ex buffer before normalize

  char* ws = (char*)d_ws;
  size_t off = 0;
  auto alloc = [&](size_t b){ size_t r = off; off = (off + b + 255) & ~(size_t)255; return r; };
  uint32_t* pWl1 = (uint32_t*)(ws + alloc((size_t)FIN*FIN/2*4));
  uint32_t* pWr1 = (uint32_t*)(ws + alloc((size_t)FIN*FIN/2*4));
  uint32_t* pWl2 = (uint32_t*)(ws + alloc((size_t)FIN*HID/2*4));
  uint32_t* pWr2 = (uint32_t*)(ws + alloc((size_t)FIN*HID/2*4));
  size_t o_xl1 = alloc((size_t)N*FIN*4);
  size_t o_xr1 = alloc((size_t)N*FIN*4);
  float* xl1 = (float*)(ws + o_xl1);
  float* xr1 = (float*)(ws + o_xr1);
  float* e1  = (float*)(ws + alloc((size_t)Et*HEADS*4));
  float* m1  = (float*)(ws + alloc((size_t)N*HEADS*4));
  float* s1  = (float*)(ws + alloc((size_t)N*HEADS*4));
  float* h1  = (float*)(ws + alloc((size_t)N*FIN*4));
  float* mu  = (float*)(ws + alloc(FIN*4));
  float* rv  = (float*)(ws + alloc(FIN*4));
  // phase 2 aliases dead phase-1 buffers
  float* xl2 = (float*)(ws + o_xl1);
  float* xr2 = (float*)(ws + o_xl1 + (size_t)N*HID*4);
  float* h2  = (float*)(ws + o_xl1 + (size_t)N*HID*8);
  float* e2  = (float*)(ws + o_xr1);
  float* m2  = (float*)(ws + o_xr1 + (((size_t)Et*4 + 255) & ~(size_t)255));
  float* s2  = m2 + N;

  auto cdiv = [](long long a, long long b){ return (unsigned)((a + b - 1) / b); };

  // --- pack weights into WMMA-B layout ---
  k_pack_b<<<cdiv(FIN*FIN/2, 256), 256, 0, stream>>>(Wl1, pWl1, FIN, FIN);
  k_pack_b<<<cdiv(FIN*FIN/2, 256), 256, 0, stream>>>(Wr1, pWr1, FIN, FIN);
  k_pack_b<<<cdiv(FIN*HID/2, 256), 256, 0, stream>>>(Wl2, pWl2, FIN, HID);
  k_pack_b<<<cdiv(FIN*HID/2, 256), 256, 0, stream>>>(Wr2, pWr2, FIN, HID);

  // --- conv1: xl1 = x@Wl1, xr1 = x@Wr1 (WMMA bf16, f32 accum) ---
  k_gemm_dual<<<cdiv(N,16), 2*(FIN/16)*32, 0, stream>>>(x, pWl1, pWr1, xl1, xr1, N, FIN);

  // --- conv1 segment softmax + aggregation ---
  k_fill<<<cdiv((long long)N*HEADS,256),256,0,stream>>>(m1, -INFINITY, (long long)N*HEADS);
  k_fill<<<cdiv((long long)N*HEADS,256),256,0,stream>>>(s1, 0.f, (long long)N*HEADS);
  k_fill<<<cdiv((long long)N*FIN,256),256,0,stream>>>(h1, 0.f, (long long)N*FIN);
  k_scores1<<<cdiv(Et,8), 256, 0, stream>>>(xl1, xr1, att1, e1, m1, ei, E, Et);
  k_exp1<<<cdiv((long long)Et*HEADS,256), 256, 0, stream>>>(e1, m1, s1, alpha, ei, E, Et);
  k_agg1<<<cdiv(Et,8), 256, 0, stream>>>(alpha, s1, xl1, h1, ei, E, Et);

  // --- bias + batchnorm + leaky ---
  k_bn_stats<<<FIN, 256, 0, stream>>>(h1, b1, mu, rv, N);
  k_bn_apply<<<cdiv((long long)N*FIN,256), 256, 0, stream>>>(h1, b1, mu, rv, gamma, beta,
                                                            (long long)N*FIN);

  // --- conv2: xl2 = h1@Wl2, xr2 = h1@Wr2 ---
  k_gemm_dual<<<cdiv(N,16), 2*(HID/16)*32, 0, stream>>>(h1, pWl2, pWr2, xl2, xr2, N, HID);

  // --- conv2 segment softmax + aggregation ---
  k_fill<<<cdiv(N,256),256,0,stream>>>(m2, -INFINITY, N);
  k_fill<<<cdiv(N,256),256,0,stream>>>(s2, 0.f, N);
  k_fill<<<cdiv((long long)N*HID,256),256,0,stream>>>(h2, 0.f, (long long)N*HID);
  k_scores2<<<cdiv(Et,8), 256, 0, stream>>>(xl2, xr2, att2, e2, m2, ei, E, Et);
  k_exp2<<<cdiv(Et,256), 256, 0, stream>>>(e2, m2, s2, ei, E, Et);
  k_agg2<<<cdiv(Et,8), 256, 0, stream>>>(e2, s2, xl2, h2, ei, E, Et);

  // --- output head + log_softmax ---
  k_out<<<cdiv(N,256), 256, 0, stream>>>(h2, b2, Wout, bout, logp, N);
}
